// HistoryAwareAnchorRouter_53291954208934
// MI455X (gfx1250) — compile-verified, matrix operand order checked by reading, not run on hardware
//
#include <hip/hip_runtime.h>
#include <math.h>

// Problem constants (match reference)
#define B_    8
#define T_    2048
#define DU_   512
#define DM_   512
#define DA_   256
#define DF_   12
#define L_    4
#define DKV_  (DA_ + DU_)     // 768, fused [k | v] output width
#define BT_   (B_ * T_)       // 16384 tokens
#define SCALE_ 16.0f          // sqrt(DA)=sqrt(256)
#define GAMMA_ 1.0f
#define EPS_   1e-6f

typedef __bf16 bf16;
typedef bf16  v16bf __attribute__((ext_vector_type(16)));
typedef bf16  v8bf  __attribute__((ext_vector_type(8)));
typedef float v8f   __attribute__((ext_vector_type(8)));

__device__ __forceinline__ float warpSum(float v) {
#pragma unroll
  for (int o = 16; o > 0; o >>= 1) v += __shfl_down(v, o, 32);
  return v;
}

// ---------------- one-time prep kernels ----------------

__global__ void k_pack_u(const float* __restrict__ u, bf16* __restrict__ uB) {
  int i = blockIdx.x * blockDim.x + threadIdx.x;
  if (i < BT_ * DU_) uB[i] = (bf16)u[i];
}

__global__ void k_pack_w(const float* __restrict__ Wk, const float* __restrict__ Wv,
                         bf16* __restrict__ wB) {
  int i = blockIdx.x * blockDim.x + threadIdx.x;
  if (i >= DKV_ * DU_) return;
  int r = i / DU_, c = i - r * DU_;
  float x = (r < DA_) ? Wk[r * DU_ + c] : Wv[(r - DA_) * DU_ + c];
  wB[i] = (bf16)x;
}

__global__ void k_usq(const float* __restrict__ u, float* __restrict__ usq) {
  __shared__ float red[256];
  int tok = blockIdx.x;
  const float* r = u + (size_t)tok * DU_;
  float a = r[threadIdx.x] * r[threadIdx.x]
          + r[threadIdx.x + 256] * r[threadIdx.x + 256];
  red[threadIdx.x] = a; __syncthreads();
  for (int s = 128; s > 0; s >>= 1) {
    if (threadIdx.x < s) red[threadIdx.x] += red[threadIdx.x + s];
    __syncthreads();
  }
  if (threadIdx.x == 0) usq[tok] = red[0];
}

// sbias[i] = gelu(desc @ w1^T + b1) @ w2^T + b2   (warp per token)
__global__ void k_sbias(const float* __restrict__ sf, const float* __restrict__ w1,
                        const float* __restrict__ b1, const float* __restrict__ w2,
                        const float* __restrict__ b2, float* __restrict__ sbias) {
  int warp = threadIdx.x >> 5, lane = threadIdx.x & 31;
  int i = blockIdx.x * 8 + warp;
  float d0 = sf[i * DF_ + (DF_ - 4)];
  float d1 = sf[i * DF_ + (DF_ - 3)];
  float d2 = sf[i * DF_ + (DF_ - 2)];
  float acc = 0.0f;
  for (int j = lane; j < DA_; j += 32) {
    float x = w1[j * 3 + 0] * d0 + w1[j * 3 + 1] * d1 + w1[j * 3 + 2] * d2 + b1[j];
    float g = 0.5f * x * (1.0f + erff(x * 0.70710678118f));   // exact gelu
    acc += g * w2[j];
  }
  acc = warpSum(acc);
  if (lane == 0) sbias[i] = acc + b2[0];
}

// ---------------- fused bf16 WMMA GEMM: kv = u @ [Wk;Wv]^T ----------------
// M = BT_ = 16384, N = DKV_ = 768, K = DU_ = 512
// Block tile 128x128, 8 waves, each wave 32(M) x 64(N) via 2x4 accumulators.
__global__ __launch_bounds__(256) void
k_gemm(const bf16* __restrict__ uB, const bf16* __restrict__ wB,
       float* __restrict__ kv) {
  int tid  = threadIdx.x;
  int wave = tid >> 5;
  int lane = tid & 31;
  int half = lane >> 4;       // 0 or 1: K-half selector per ISA layout
  int l16  = lane & 15;       // row (A) / col (B) within tile
  int waveN = wave & 1;
  int waveM = wave >> 1;
  int rowBase = blockIdx.y * 128 + waveM * 32;
  int colBase = blockIdx.x * 128 + waveN * 64;

  v8f acc[2][4] = {};

  for (int k0 = 0; k0 < DU_; k0 += 32) {
    v16bf a[2], b[4];
#pragma unroll
    for (int mt = 0; mt < 2; ++mt) {
      const bf16* p = uB + (size_t)(rowBase + mt * 16 + l16) * DU_ + k0 + half * 8;
      v8bf c0 = *(const v8bf*)p;           // K = k0 + 8*half    .. +8
      v8bf c1 = *(const v8bf*)(p + 16);    // K = k0+16+8*half   .. +8
#pragma unroll
      for (int i = 0; i < 8; ++i) { a[mt][i] = c0[i]; a[mt][i + 8] = c1[i]; }
    }
#pragma unroll
    for (int nt = 0; nt < 4; ++nt) {
      const bf16* p = wB + (size_t)(colBase + nt * 16 + l16) * DU_ + k0 + half * 8;
      v8bf c0 = *(const v8bf*)p;
      v8bf c1 = *(const v8bf*)(p + 16);
#pragma unroll
      for (int i = 0; i < 8; ++i) { b[nt][i] = c0[i]; b[nt][i + 8] = c1[i]; }
    }
#pragma unroll
    for (int mt = 0; mt < 2; ++mt)
#pragma unroll
      for (int nt = 0; nt < 4; ++nt)
        acc[mt][nt] = __builtin_amdgcn_wmma_f32_16x16x32_bf16(
            false, a[mt], false, b[nt], (short)0, acc[mt][nt], false, false);
  }

  // C/D layout: lane%16 = N; VGPR r -> M = r + 8*(lane/16)
#pragma unroll
  for (int mt = 0; mt < 2; ++mt)
#pragma unroll
    for (int nt = 0; nt < 4; ++nt)
#pragma unroll
      for (int r = 0; r < 8; ++r) {
        int m = rowBase + mt * 16 + half * 8 + r;
        int n = colBase + nt * 16 + l16;
        kv[(size_t)m * DKV_ + n] = acc[mt][nt][r];
      }
}

// ---------------- per-stage kernels ----------------

__global__ void k_init(const float* __restrict__ m0, float* __restrict__ mem,
                       float* __restrict__ prev) {
  int i = blockIdx.x * blockDim.x + threadIdx.x;
  if (i < BT_) prev[i] = 0.0f;
  if (i < B_ * DM_) mem[i] = m0[i];
}

// q[b] = Wq @ memory[b]   (block per batch, thread per output)
__global__ void k_q(const float* __restrict__ Wq, const float* __restrict__ mem,
                    float* __restrict__ q) {
  __shared__ float msh[DM_];
  int b = blockIdx.x, tid = threadIdx.x;
  msh[tid]       = mem[b * DM_ + tid];
  msh[tid + 256] = mem[b * DM_ + tid + 256];
  __syncthreads();
  const float* row = Wq + (size_t)tid * DM_;
  float a = 0.0f;
  for (int i = 0; i < DM_; ++i) a += row[i] * msh[i];
  q[b * DA_ + tid] = a;
}

// logits[b,t] = (k.q/SCALE + sbias - gamma*sim@prev) / temp   (warp per token)
__global__ void k_scores(const float* __restrict__ kv, const float* __restrict__ q,
                         const float* __restrict__ sbias, const float* __restrict__ sim,
                         const float* __restrict__ prev, const float* __restrict__ log_temp,
                         float* __restrict__ logits, int stage) {
  int b = blockIdx.y;
  int warp = threadIdx.x >> 5, lane = threadIdx.x & 31;
  int t = blockIdx.x * 8 + warp;
  const float* krow = kv + (size_t)(b * T_ + t) * DKV_;   // first DA_ cols = k
  const float* qb = q + b * DA_;
  float kq = 0.0f;
  for (int a = lane; a < DA_; a += 32) kq += krow[a] * qb[a];
  float sd = 0.0f;
  if (stage > 0) {
    const float* srow = sim + ((size_t)b * T_ + t) * T_;
    const float* pb = prev + b * T_;
    for (int s = lane; s < T_; s += 32) sd += srow[s] * pb[s];
  }
  kq = warpSum(kq); sd = warpSum(sd);
  if (lane == 0) {
    float temp = fminf(fmaxf(expf(log_temp[0]), 0.1f), 10.0f);
    logits[b * T_ + t] = (kq / SCALE_ + sbias[b * T_ + t] - GAMMA_ * sd) / temp;
  }
}

__global__ __launch_bounds__(1024) void
k_softmax(const float* __restrict__ logits, float* __restrict__ y) {
  __shared__ float red[1024];
  int b = blockIdx.x, tid = threadIdx.x;
  float l0 = logits[b * T_ + tid], l1 = logits[b * T_ + tid + 1024];
  red[tid] = fmaxf(l0, l1); __syncthreads();
  for (int s = 512; s > 0; s >>= 1) {
    if (tid < s) red[tid] = fmaxf(red[tid], red[tid + s]);
    __syncthreads();
  }
  float mx = red[0]; __syncthreads();
  float e0 = expf(l0 - mx), e1 = expf(l1 - mx);
  red[tid] = e0 + e1; __syncthreads();
  for (int s = 512; s > 0; s >>= 1) {
    if (tid < s) red[tid] += red[tid + s];
    __syncthreads();
  }
  float S = red[0];
  y[b * T_ + tid]        = e0 / S;
  y[b * T_ + tid + 1024] = e1 / S;
}

// overlap[b,t] = sim[b,t,:] . y[b,:]   (warp per token; sim is L2-resident)
__global__ void k_overlap(const float* __restrict__ sim, const float* __restrict__ y,
                          float* __restrict__ ovl) {
  int b = blockIdx.y;
  int warp = threadIdx.x >> 5, lane = threadIdx.x & 31;
  int t = blockIdx.x * 8 + warp;
  const float* srow = sim + ((size_t)b * T_ + t) * T_;
  const float* yb = y + b * T_;
  float a = 0.0f;
  for (int s = lane; s < T_; s += 32) a += srow[s] * yb[s];
  a = warpSum(a);
  if (lane == 0) ovl[b * T_ + t] = a;
}

// penalty + re-softmax + fused T-reductions (ysum, entropy, wsq)
__global__ __launch_bounds__(1024) void
k_stage2(const float* __restrict__ logits, const float* __restrict__ ovl,
         const float* __restrict__ usq, float* __restrict__ y,
         float* __restrict__ stats) {
  __shared__ float red[1024];
  int b = blockIdx.x, tid = threadIdx.x;
  int t0 = tid, t1 = tid + 1024;
  float o0 = ovl[b * T_ + t0], o1 = ovl[b * T_ + t1];
  red[tid] = fmaxf(o0, o1); __syncthreads();
  for (int s = 512; s > 0; s >>= 1) {
    if (tid < s) red[tid] = fmaxf(red[tid], red[tid + s]);
    __syncthreads();
  }
  float mo = fmaxf(red[0], EPS_); __syncthreads();
  float l0 = logits[b * T_ + t0] - o0 / mo;
  float l1 = logits[b * T_ + t1] - o1 / mo;
  red[tid] = fmaxf(l0, l1); __syncthreads();
  for (int s = 512; s > 0; s >>= 1) {
    if (tid < s) red[tid] = fmaxf(red[tid], red[tid + s]);
    __syncthreads();
  }
  float mx = red[0]; __syncthreads();
  float e0 = expf(l0 - mx), e1 = expf(l1 - mx);
  red[tid] = e0 + e1; __syncthreads();
  for (int s = 512; s > 0; s >>= 1) {
    if (tid < s) red[tid] += red[tid + s];
    __syncthreads();
  }
  float S = red[0]; __syncthreads();
  float y0 = e0 / S, y1 = e1 / S;
  y[b * T_ + t0] = y0; y[b * T_ + t1] = y1;
  // ysum
  red[tid] = y0 + y1; __syncthreads();
  for (int s = 512; s > 0; s >>= 1) {
    if (tid < s) red[tid] += red[tid + s];
    __syncthreads();
  }
  float ys = fmaxf(red[0], EPS_); __syncthreads();
  // entropy of y/ysum
  float yn0 = y0 / ys, yn1 = y1 / ys;
  red[tid] = -(yn0 * logf(fmaxf(yn0, EPS_)) + yn1 * logf(fmaxf(yn1, EPS_)));
  __syncthreads();
  for (int s = 512; s > 0; s >>= 1) {
    if (tid < s) red[tid] += red[tid + s];
    __syncthreads();
  }
  float H = red[0]; __syncthreads();
  // wsq = sum y * |u|^2
  red[tid] = y0 * usq[b * T_ + t0] + y1 * usq[b * T_ + t1]; __syncthreads();
  for (int s = 512; s > 0; s >>= 1) {
    if (tid < s) red[tid] += red[tid + s];
    __syncthreads();
  }
  if (tid == 0) {
    stats[b * 16 + 0] = ys;
    stats[b * 16 + 1] = H;
    stats[b * 16 + 2] = red[0];   // wsq
  }
}

// z[b,d] = sum_t y*v ; centroid[b,d] = sum_t y*u ; also writes output slice
__global__ void k_zc(const float* __restrict__ u, const float* __restrict__ kv,
                     const float* __restrict__ y, float* __restrict__ zws,
                     float* __restrict__ cent, float* __restrict__ out, int stage) {
  __shared__ float rz[256], rc[256];
  int b = blockIdx.y;
  int dcol = threadIdx.x & 63;
  int d = blockIdx.x * 64 + dcol;
  int tg = threadIdx.x >> 6;           // 4 token groups
  float za = 0.0f, ca = 0.0f;
  for (int t = tg; t < T_; t += 4) {
    float yv = y[b * T_ + t];
    za += yv * kv[(size_t)(b * T_ + t) * DKV_ + DA_ + d];   // v columns
    ca += yv * u[(size_t)(b * T_ + t) * DU_ + d];
  }
  rz[threadIdx.x] = za; rc[threadIdx.x] = ca; __syncthreads();
  if (tg == 0) {
    float z = rz[dcol] + rz[dcol + 64] + rz[dcol + 128] + rz[dcol + 192];
    float c = rc[dcol] + rc[dcol + 64] + rc[dcol + 128] + rc[dcol + 192];
    zws[b * DU_ + d] = z;
    cent[b * DU_ + d] = c;
    out[((size_t)b * L_ + stage) * DU_ + d] = z;
  }
}

// spread + compact + final 4-feature stats vector
__global__ __launch_bounds__(1024) void
k_spread(const float* __restrict__ u, const float* __restrict__ y,
         const float* __restrict__ cent, float* __restrict__ stats) {
  __shared__ float cen[DU_];
  __shared__ float red[1024];
  __shared__ float wp[32];
  int b = blockIdx.x, tid = threadIdx.x;
  float ys = fmaxf(stats[b * 16 + 0], EPS_);
  float cs = 0.0f;
  if (tid < DU_) {
    float cv = cent[b * DU_ + tid];
    cen[tid] = cv / ys;
    cs = cv * cv;
  }
  red[tid] = cs; __syncthreads();
  for (int s = 512; s > 0; s >>= 1) {
    if (tid < s) red[tid] += red[tid + s];
    __syncthreads();
  }
  float csq = red[0];
  int warp = tid >> 5, lane = tid & 31;
  float acc = 0.0f;
  for (int t = warp; t < T_; t += 32) {
    const float* ur = u + (size_t)(b * T_ + t) * DU_;
    float sq = 0.0f;
    for (int d = lane; d < DU_; d += 32) {
      float df = ur[d] - cen[d];
      sq += df * df;
    }
    sq = warpSum(sq);
    if (lane == 0) acc += y[b * T_ + t] * sqrtf(sq);
  }
  if (lane == 0) wp[warp] = acc;
  __syncthreads();
  if (tid == 0) {
    float sp = 0.0f;
    for (int w = 0; w < 32; ++w) sp += wp[w];
    float H = stats[b * 16 + 1], wsq = stats[b * 16 + 2];
    stats[b * 16 + 4] = ys / (float)T_;                              // coverage
    stats[b * 16 + 5] = H;                                           // entropy
    stats[b * 16 + 6] = sp / ys;                                     // spread
    stats[b * 16 + 7] = 2.0f * (ys * wsq - csq) / fmaxf(ys * ys, EPS_); // compact
  }
}

// GRU memory update (block per batch, thread per memory unit)
__global__ __launch_bounds__(512) void
k_gru(const float* __restrict__ Wih, const float* __restrict__ bih,
      const float* __restrict__ Whh, const float* __restrict__ bhh,
      const float* __restrict__ zws, const float* __restrict__ stats,
      float* __restrict__ mem) {
  __shared__ float xsh[DU_ + 8];
  __shared__ float msh[DM_];
  int b = blockIdx.x, j = threadIdx.x;
  msh[j] = mem[b * DM_ + j];
  xsh[j] = zws[b * DU_ + j];
  if (j < 4) xsh[DU_ + j] = stats[b * 16 + 4 + j];
  __syncthreads();
  const int NX = DU_ + 4;
  float gir = bih[j], giz = bih[DM_ + j], gin = bih[2 * DM_ + j];
  const float* r0 = Wih + (size_t)j * NX;
  const float* r1 = Wih + (size_t)(DM_ + j) * NX;
  const float* r2 = Wih + (size_t)(2 * DM_ + j) * NX;
  for (int i = 0; i < NX; ++i) {
    float xv = xsh[i];
    gir += r0[i] * xv; giz += r1[i] * xv; gin += r2[i] * xv;
  }
  float ghr = bhh[j], ghz = bhh[DM_ + j], ghn = bhh[2 * DM_ + j];
  const float* h0 = Whh + (size_t)j * DM_;
  const float* h1 = Whh + (size_t)(DM_ + j) * DM_;
  const float* h2 = Whh + (size_t)(2 * DM_ + j) * DM_;
  for (int i = 0; i < DM_; ++i) {
    float mv = msh[i];
    ghr += h0[i] * mv; ghz += h1[i] * mv; ghn += h2[i] * mv;
  }
  float rg = 1.0f / (1.0f + expf(-(gir + ghr)));
  float zg = 1.0f / (1.0f + expf(-(giz + ghz)));
  float ng = tanhf(gin + rg * ghn);
  mem[b * DM_ + j] = (1.0f - zg) * ng + zg * msh[j];
}

__global__ void k_prev(const float* __restrict__ y, float* __restrict__ prev) {
  int i = blockIdx.x * blockDim.x + threadIdx.x;
  prev[i] += y[i];
}

// ---------------- launcher ----------------

extern "C" void kernel_launch(void* const* d_in, const int* in_sizes, int n_in,
                              void* d_out, int out_size, void* d_ws, size_t ws_size,
                              hipStream_t stream) {
  (void)in_sizes; (void)n_in; (void)out_size; (void)ws_size;
  const float* u   = (const float*)d_in[0];
  const float* sf  = (const float*)d_in[1];
  const float* sim = (const float*)d_in[2];
  const float* m0  = (const float*)d_in[3];
  const float* Wq  = (const float*)d_in[4];
  const float* Wk  = (const float*)d_in[5];
  const float* Wv  = (const float*)d_in[6];
  const float* sw1 = (const float*)d_in[7];
  const float* sb1 = (const float*)d_in[8];
  const float* sw2 = (const float*)d_in[9];
  const float* sb2 = (const float*)d_in[10];
  const float* lt  = (const float*)d_in[11];
  const float* Wih = (const float*)d_in[12];
  const float* bih = (const float*)d_in[13];
  const float* Whh = (const float*)d_in[14];
  const float* bhh = (const float*)d_in[15];
  float* out = (float*)d_out;

  char* ws = (char*)d_ws;
  size_t off = 0;
  auto alloc = [&](size_t bytes) -> char* {
    char* p = ws + off;
    off = (off + bytes + 255) & ~(size_t)255;
    return p;
  };
  bf16*  uB     = (bf16*) alloc((size_t)BT_ * DU_ * 2);
  bf16*  wB     = (bf16*) alloc((size_t)DKV_ * DU_ * 2);
  float* kv     = (float*)alloc((size_t)BT_ * DKV_ * 4);
  float* usq    = (float*)alloc((size_t)BT_ * 4);
  float* sbias  = (float*)alloc((size_t)BT_ * 4);
  float* logits = (float*)alloc((size_t)BT_ * 4);
  float* yb     = (float*)alloc((size_t)BT_ * 4);
  float* prev   = (float*)alloc((size_t)BT_ * 4);
  float* ovl    = (float*)alloc((size_t)BT_ * 4);
  float* mem    = (float*)alloc((size_t)B_ * DM_ * 4);
  float* qv     = (float*)alloc((size_t)B_ * DA_ * 4);
  float* zws    = (float*)alloc((size_t)B_ * DU_ * 4);
  float* cent   = (float*)alloc((size_t)B_ * DU_ * 4);
  float* stats  = (float*)alloc((size_t)B_ * 16 * 4);

  // one-time prep
  k_pack_u<<<(BT_ * DU_) / 256, 256, 0, stream>>>(u, uB);
  k_pack_w<<<(DKV_ * DU_) / 256, 256, 0, stream>>>(Wk, Wv, wB);
  k_usq<<<BT_, 256, 0, stream>>>(u, usq);
  k_sbias<<<BT_ / 8, 256, 0, stream>>>(sf, sw1, sb1, sw2, sb2, sbias);
  k_gemm<<<dim3(DKV_ / 128, BT_ / 128), 256, 0, stream>>>(uB, wB, kv);
  k_init<<<BT_ / 256, 256, 0, stream>>>(m0, mem, prev);

  for (int stage = 0; stage < L_; ++stage) {
    k_q      <<<B_, 256, 0, stream>>>(Wq, mem, qv);
    k_scores <<<dim3(T_ / 8, B_), 256, 0, stream>>>(kv, qv, sbias, sim, prev, lt, logits, stage);
    k_softmax<<<B_, 1024, 0, stream>>>(logits, yb);
    k_overlap<<<dim3(T_ / 8, B_), 256, 0, stream>>>(sim, yb, ovl);
    k_stage2 <<<B_, 1024, 0, stream>>>(logits, ovl, usq, yb, stats);
    k_zc     <<<dim3(DU_ / 64, B_), 256, 0, stream>>>(u, kv, yb, zws, cent, out, stage);
    k_spread <<<B_, 1024, 0, stream>>>(u, yb, cent, stats);
    k_gru    <<<B_, 512, 0, stream>>>(Wih, bih, Whh, bhh, zws, stats, mem);
    k_prev   <<<BT_ / 256, 256, 0, stream>>>(yb, prev);
  }
}
